// Hydra_16767552324181
// MI455X (gfx1250) — matrix operand order repaired and checked by reading, not compiled, not run on hardware
//
#include <hip/hip_runtime.h>
#include <hip/hip_bf16.h>
#include <math.h>

typedef __attribute__((ext_vector_type(16))) _Float16 v16h;
typedef __attribute__((ext_vector_type(8)))  float    v8f;

#define NT   65536   // B*T tokens (B=128, T=512)
#define TT   512
#define DDIM 128
#define NHEAD 4
#define DH   32
#define NEXP 4
#define EDIM 256
#define NLAY 4

__device__ __forceinline__ v8f wmma_f16(v16h a, v16h b, v8f c) {
  return __builtin_amdgcn_wmma_f32_16x16x32_f16(false, a, false, b, (short)0, c,
                                                false, false);
}

// ---- async global->LDS copy (CDNA5), with safe fallback --------------------
#if __has_builtin(__builtin_amdgcn_global_load_async_to_lds_b32) && \
    __has_builtin(__builtin_amdgcn_s_wait_asynccnt)
#define USE_ASYNC_LDS 1
typedef __attribute__((address_space(1))) int as1_int;
typedef __attribute__((address_space(3))) int as3_int;
__device__ __forceinline__ void async_copy_b32(const float* g, float* l) {
  __builtin_amdgcn_global_load_async_to_lds_b32(
      (as1_int*)(void*)const_cast<float*>(g),
      (as3_int*)(void*)l, 0, 0);
}
#endif

// ---- max-reduction across the 16 lanes of each lane-row (butterfly xor) ----
#if __has_builtin(__builtin_amdgcn_permlane16_b32)
#define PERML16(v, lo, hi) __builtin_amdgcn_permlane16_b32((v), (v), (int)(lo), (int)(hi), false, false)
#define HAVE_PERML16 1
#elif __has_builtin(__builtin_amdgcn_permlane16)
#define PERML16(v, lo, hi) __builtin_amdgcn_permlane16((v), (v), (int)(lo), (int)(hi), false, false)
#define HAVE_PERML16 1
#endif

__device__ __forceinline__ float rowmax16(float x) {
#ifdef HAVE_PERML16
  // selectors: nibble i = i ^ m   (lanes 0-7 in lo, 8-15 in hi; applied per 16-lane row)
  int xi;
  xi = __float_as_int(x); x = fmaxf(x, __int_as_float(PERML16(xi, 0x67452301u, 0xEFCDAB89u))); // xor 1
  xi = __float_as_int(x); x = fmaxf(x, __int_as_float(PERML16(xi, 0x54761032u, 0xDCFE98BAu))); // xor 2
  xi = __float_as_int(x); x = fmaxf(x, __int_as_float(PERML16(xi, 0x32107654u, 0xBA98FEDCu))); // xor 4
  xi = __float_as_int(x); x = fmaxf(x, __int_as_float(PERML16(xi, 0xFEDCBA98u, 0x76543210u))); // xor 8
#else
  #pragma unroll
  for (int off = 1; off < 16; off <<= 1) x = fmaxf(x, __shfl_xor(x, off, 32));
#endif
  return x;
}

// ---------------------------------------------------------------------------
// Generic tiled GEMM: C[M,N] = act(A[M,K] @ W[K,N] + bias) * rowscale (+=)
// Block = 128 threads (4 waves). Each wave owns a 16-row tile; the four waves
// share a 32(K) x 64(N) f32 W slab, staged in LDS via async global->LDS copy.
// ---------------------------------------------------------------------------
template <int NTILES>
__global__ void gemm_wmma(const float* __restrict__ A, const float* __restrict__ W,
                          const float* __restrict__ bias,
                          const float* __restrict__ rowscale, int rs_stride,
                          float* __restrict__ C,
                          int Kd, int N, int act, int accum)
{
  const int tid  = threadIdx.x;
  const int wave = tid >> 5;
  const int lane = tid & 31;
  const int rt   = blockIdx.x * 4 + wave;   // 16-row tile index (grid covers M)
  const int n0   = blockIdx.y * 64;
  const int hi   = lane >> 4;        // lane half
  const int mrow = lane & 15;        // A row within tile
  const int ncol = lane & 15;        // B/C column within tile
  const int koff = hi << 3;          // A K-group offset {0,8}
  const int kb16 = hi << 4;          // unused; B uses koff, as A does
  const int crow = hi << 3;          // C row base {0,8}

  __shared__ float sbf[32 * 64];     // W slab: [k][n], f32

  v8f acc[NTILES];
  #pragma unroll
  for (int j = 0; j < NTILES; ++j) acc[j] = (v8f){0.f,0.f,0.f,0.f,0.f,0.f,0.f,0.f};

  const float* arow = A + (size_t)(rt * 16 + mrow) * Kd;

  for (int ks = 0; ks < Kd; ks += 32) {
    // ---- cooperative W slab fill (coalesced: consecutive tid -> consecutive n)
    __syncthreads();
    #pragma unroll
    for (int r = 0; r < 16; ++r) {
      const int idx = tid + (r << 7);   // 0..2047
      const int k   = idx >> 6;
      const int n   = idx & 63;
      if (n0 + n < N) {
#ifdef USE_ASYNC_LDS
        async_copy_b32(&W[(size_t)(ks + k) * N + (n0 + n)], &sbf[idx]);
#else
        sbf[idx] = W[(size_t)(ks + k) * N + (n0 + n)];
#endif
      }
    }
#ifdef USE_ASYNC_LDS
    __builtin_amdgcn_s_wait_asynccnt(0);
#endif
    __syncthreads();
    if (ks + 32 < Kd)   // speculative prefetch of next slab
      __builtin_prefetch(W + (size_t)(ks + 32) * N + n0 + (tid & 63), 0, 1);

    // ---- A tile in A-layout (two 32B-aligned float4 pairs per lane)
    const float4 a0 = *(const float4*)(arow + ks + koff);
    const float4 a1 = *(const float4*)(arow + ks + koff + 4);
    const float4 a2 = *(const float4*)(arow + ks + 16 + koff);
    const float4 a3 = *(const float4*)(arow + ks + 16 + koff + 4);
    v16h a;
    a[0]=(_Float16)a0.x;  a[1]=(_Float16)a0.y;  a[2]=(_Float16)a0.z;  a[3]=(_Float16)a0.w;
    a[4]=(_Float16)a1.x;  a[5]=(_Float16)a1.y;  a[6]=(_Float16)a1.z;  a[7]=(_Float16)a1.w;
    a[8]=(_Float16)a2.x;  a[9]=(_Float16)a2.y;  a[10]=(_Float16)a2.z; a[11]=(_Float16)a2.w;
    a[12]=(_Float16)a3.x; a[13]=(_Float16)a3.y; a[14]=(_Float16)a3.z; a[15]=(_Float16)a3.w;

    #pragma unroll
    for (int j = 0; j < NTILES; ++j) {
      const int jn = j * 16 + ncol;
      v16h bm;
      #pragma unroll
      for (int i = 0; i < 16; ++i) bm[i] = (_Float16)sbf[(((i < 8) ? (koff + i) : (8 + koff + i)) << 6) + jn];
      acc[j] = wmma_f16(a, bm, acc[j]);
    }
  }

  #pragma unroll
  for (int j = 0; j < NTILES; ++j) {
    #pragma unroll
    for (int v = 0; v < 8; ++v) {
      const int row = rt * 16 + crow + v;
      const int col = n0 + j * 16 + ncol;
      float val = acc[j][v];
      if (bias)     val += bias[col];
      if (act)      val = 0.5f * val * (1.f + erff(val * 0.70710678118f));
      if (rowscale) val *= rowscale[(size_t)row * rs_stride];
      const size_t idx = (size_t)row * N + col;
      if (accum) C[idx] += val; else C[idx] = val;
    }
  }
}

// ---------------------------------------------------------------------------
// Flash attention for MLA: one wave per (b, head, 16-query tile), T=512, DH=32
// Row sums of P come from an extra WMMA against an all-ones B matrix (every
// lane then holds its row's sum); row max via permlane16 butterflies.
// ---------------------------------------------------------------------------
__global__ void mla_attn(const float* __restrict__ Q, const float* __restrict__ K,
                         const float* __restrict__ V, float* __restrict__ O)
{
  const int wave = threadIdx.x >> 5;
  const int lane = threadIdx.x & 31;
  const int qt   = blockIdx.x & 31;                 // query tile 0..31
  const int bh   = (blockIdx.x >> 5) * 4 + wave;    // 0..511
  const int b    = bh >> 2;
  const int hh   = bh & 3;

  const int hi   = lane >> 4;
  const int mrow = lane & 15;
  const int ncol = lane & 15;
  const int koff = hi << 3;
  const int kb16 = hi << 4;
  const int crow = hi << 3;

  __shared__ _Float16 pb[4][16 * 32];
  _Float16* pl = pb[wave];

  const float SCALE = 0.1767766953f;   // 1/sqrt(32)

  // Q tile in A-layout (16 queries x 32 dims), pre-scaled by 1/sqrt(DH)
  const float* qrow = Q + (size_t)(b * TT + qt * 16 + mrow) * DDIM + hh * DH;
  const float4 q0 = *(const float4*)(qrow + koff);
  const float4 q1 = *(const float4*)(qrow + koff + 4);
  const float4 q2 = *(const float4*)(qrow + 16 + koff);
  const float4 q3 = *(const float4*)(qrow + 16 + koff + 4);
  v16h aq;
  aq[0]=(_Float16)(q0.x*SCALE);  aq[1]=(_Float16)(q0.y*SCALE);
  aq[2]=(_Float16)(q0.z*SCALE);  aq[3]=(_Float16)(q0.w*SCALE);
  aq[4]=(_Float16)(q1.x*SCALE);  aq[5]=(_Float16)(q1.y*SCALE);
  aq[6]=(_Float16)(q1.z*SCALE);  aq[7]=(_Float16)(q1.w*SCALE);
  aq[8]=(_Float16)(q2.x*SCALE);  aq[9]=(_Float16)(q2.y*SCALE);
  aq[10]=(_Float16)(q2.z*SCALE); aq[11]=(_Float16)(q2.w*SCALE);
  aq[12]=(_Float16)(q3.x*SCALE); aq[13]=(_Float16)(q3.y*SCALE);
  aq[14]=(_Float16)(q3.z*SCALE); aq[15]=(_Float16)(q3.w*SCALE);

  v16h ones;
  #pragma unroll
  for (int i = 0; i < 16; ++i) ones[i] = (_Float16)1.0f;

  float runm[8], runl[8], al8[8];
  #pragma unroll
  for (int v = 0; v < 8; ++v) { runm[v] = -3.0e38f; runl[v] = 0.f; }
  v8f o0 = (v8f){0.f,0.f,0.f,0.f,0.f,0.f,0.f,0.f};
  v8f o1 = o0;

  // pointer-increment bases (advance by 32 keys per step)
  const float* kp0 = K + (size_t)(b * TT + ncol) * DDIM + hh * DH + koff;
  const float* kp1 = kp0 + (size_t)16 * DDIM;
  const float* vpb = V + (size_t)(b * TT + koff) * DDIM + hh * DH;
  const size_t kstride = (size_t)32 * DDIM;

  const int ksteps = (qt * 16 + 15) / 32 + 1;       // causal upper bound
  for (int kt = 0; kt < ksteps; ++kt) {
    const int key0 = kt * 32;
    // ---- K^T tiles in B-layout: b[i] = K[key][dim koff + i (+8 if i>=8)]
    float4 k00 = *(const float4*)(kp0);     float4 k01 = *(const float4*)(kp0 + 4);
    float4 k02 = *(const float4*)(kp0 + 16); float4 k03 = *(const float4*)(kp0 + 20);
    float4 k10 = *(const float4*)(kp1);     float4 k11 = *(const float4*)(kp1 + 4);
    float4 k12 = *(const float4*)(kp1 + 16); float4 k13 = *(const float4*)(kp1 + 20);
    v16h bk0, bk1;
    bk0[0]=(_Float16)k00.x;  bk0[1]=(_Float16)k00.y;  bk0[2]=(_Float16)k00.z;  bk0[3]=(_Float16)k00.w;
    bk0[4]=(_Float16)k01.x;  bk0[5]=(_Float16)k01.y;  bk0[6]=(_Float16)k01.z;  bk0[7]=(_Float16)k01.w;
    bk0[8]=(_Float16)k02.x;  bk0[9]=(_Float16)k02.y;  bk0[10]=(_Float16)k02.z; bk0[11]=(_Float16)k02.w;
    bk0[12]=(_Float16)k03.x; bk0[13]=(_Float16)k03.y; bk0[14]=(_Float16)k03.z; bk0[15]=(_Float16)k03.w;
    bk1[0]=(_Float16)k10.x;  bk1[1]=(_Float16)k10.y;  bk1[2]=(_Float16)k10.z;  bk1[3]=(_Float16)k10.w;
    bk1[4]=(_Float16)k11.x;  bk1[5]=(_Float16)k11.y;  bk1[6]=(_Float16)k11.z;  bk1[7]=(_Float16)k11.w;
    bk1[8]=(_Float16)k12.x;  bk1[9]=(_Float16)k12.y;  bk1[10]=(_Float16)k12.z; bk1[11]=(_Float16)k12.w;
    bk1[12]=(_Float16)k13.x; bk1[13]=(_Float16)k13.y; bk1[14]=(_Float16)k13.z; bk1[15]=(_Float16)k13.w;
    kp0 += kstride; kp1 += kstride;

    v8f zero8 = (v8f){0.f,0.f,0.f,0.f,0.f,0.f,0.f,0.f};
    v8f s0 = wmma_f16(aq, bk0, zero8);
    v8f s1 = wmma_f16(aq, bk1, zero8);

    // ---- online softmax: mask, row-max (permlane butterfly), exp, stash P
    #pragma unroll
    for (int v = 0; v < 8; ++v) {
      const int qg  = qt * 16 + crow + v;
      float x0 = s0[v] + (((key0 + ncol)      > qg) ? -1.0e9f : 0.f);
      float x1 = s1[v] + (((key0 + 16 + ncol) > qg) ? -1.0e9f : 0.f);
      const float mx = rowmax16(fmaxf(x0, x1));
      const float nm = fmaxf(runm[v], mx);
      al8[v] = __expf(runm[v] - nm);
      runm[v] = nm;
      x0 = __expf(x0 - nm);
      x1 = __expf(x1 - nm);
      pl[(crow + v) * 32 + ncol]      = (_Float16)x0;
      pl[(crow + v) * 32 + 16 + ncol] = (_Float16)x1;
    }
    asm volatile("s_wait_dscnt 0" ::: "memory");
    v16h pa;
    #pragma unroll
    for (int i = 0; i < 8; ++i) {
      pa[i]     = pl[mrow * 32 + koff + i];
      pa[8 + i] = pl[mrow * 32 + 16 + koff + i];
    }
    asm volatile("s_wait_dscnt 0" ::: "memory");

    // ---- row sums of P via WMMA with all-ones B (replicated to every lane)
    v8f rs = wmma_f16(pa, ones, zero8);
    #pragma unroll
    for (int v = 0; v < 8; ++v) {
      runl[v] = runl[v] * al8[v] + rs[v];
      o0[v] *= al8[v];
      o1[v] *= al8[v];
    }

    // ---- V tiles in B-layout: b[i] = V[key0 + koff + i (+8 if i>=8)][dim ncol(+16)]
    v16h bv0, bv1;
    #pragma unroll
    for (int i = 0; i < 16; ++i) {
      const float* vp = vpb + (size_t)((i < 8) ? i : (i + 8)) * DDIM;
      bv0[i] = (_Float16)vp[ncol];
      bv1[i] = (_Float16)vp[16 + ncol];
    }
    vpb += kstride;
    o0 = wmma_f16(pa, bv0, o0);
    o1 = wmma_f16(pa, bv1, o1);
  }
  #pragma unroll
  for (int v = 0; v < 8; ++v) {
    const float inv = 1.f / runl[v];
    float* orow = O + (size_t)(b * TT + qt * 16 + crow + v) * DDIM + hh * DH;
    orow[ncol]      = o0[v] * inv;
    orow[16 + ncol] = o1[v] * inv;
  }
}

// ---------------------------------------------------------------------------
// h[tok] += pos_emb[tok % T]
// ---------------------------------------------------------------------------
__global__ void pos_add(float* __restrict__ h, const float* __restrict__ pos)
{
  const int tok = blockIdx.x, d = threadIdx.x;
  h[(size_t)tok * DDIM + d] += pos[(size_t)(tok & (TT - 1)) * DDIM + d];
}

// ---------------------------------------------------------------------------
// out = LayerNorm(base + res) * g + be   (res may be null)
// ---------------------------------------------------------------------------
__global__ void add_ln(const float* __restrict__ base, const float* __restrict__ res,
                       const float* __restrict__ g, const float* __restrict__ be,
                       float* __restrict__ out)
{
  const int tok = blockIdx.x, d = threadIdx.x;
  __shared__ float red[DDIM];
  float v = base[(size_t)tok * DDIM + d];
  if (res) v += res[(size_t)tok * DDIM + d];
  red[d] = v; __syncthreads();
  for (int s = 64; s > 0; s >>= 1) { if (d < s) red[d] += red[d + s]; __syncthreads(); }
  const float mean = red[0] * (1.f / DDIM);
  __syncthreads();
  const float diff = v - mean;
  red[d] = diff * diff; __syncthreads();
  for (int s = 64; s > 0; s >>= 1) { if (d < s) red[d] += red[d + s]; __syncthreads(); }
  const float var = red[0] * (1.f / DDIM);
  out[(size_t)tok * DDIM + d] = g[d] * diff * rsqrtf(var + 1e-6f) + be[d];
}

// ---------------------------------------------------------------------------
// router: w[tok,e] = softmax_e(h @ Wr + br)
// ---------------------------------------------------------------------------
__global__ void router_k(const float* __restrict__ h, const float* __restrict__ Wr,
                         const float* __restrict__ br, float* __restrict__ w)
{
  const int tok = blockIdx.x, d = threadIdx.x;
  __shared__ float red[NEXP][DDIM];
  const float hv = h[(size_t)tok * DDIM + d];
  #pragma unroll
  for (int e = 0; e < NEXP; ++e) red[e][d] = hv * Wr[(size_t)d * NEXP + e];
  __syncthreads();
  for (int s = 64; s > 0; s >>= 1) {
    if (d < s) {
      #pragma unroll
      for (int e = 0; e < NEXP; ++e) red[e][d] += red[e][d + s];
    }
    __syncthreads();
  }
  if (d == 0) {
    float lg[NEXP], mx = -3.0e38f;
    #pragma unroll
    for (int e = 0; e < NEXP; ++e) { lg[e] = red[e][0] + br[e]; mx = fmaxf(mx, lg[e]); }
    float sum = 0.f;
    #pragma unroll
    for (int e = 0; e < NEXP; ++e) { lg[e] = __expf(lg[e] - mx); sum += lg[e]; }
    const float inv = 1.f / sum;
    #pragma unroll
    for (int e = 0; e < NEXP; ++e) w[(size_t)tok * NEXP + e] = lg[e] * inv;
  }
}

// ---------------------------------------------------------------------------
// head: out[tok,j] = ln_h[tok] @ W_head[:,j] + b_head[j]   (j = 0..4)
// ---------------------------------------------------------------------------
__global__ void head_k(const float* __restrict__ lnh, const float* __restrict__ Wh,
                       const float* __restrict__ bh, float* __restrict__ out)
{
  const int tok = blockIdx.x, d = threadIdx.x;
  __shared__ float sh[DDIM];
  sh[d] = lnh[(size_t)tok * DDIM + d];
  __syncthreads();
  if (d < 5) {
    float acc = bh[d];
    for (int i = 0; i < DDIM; ++i) acc += sh[i] * Wh[(size_t)i * 5 + d];
    out[(size_t)tok * 5 + d] = acc;
  }
}

// ---------------------------------------------------------------------------
extern "C" void kernel_launch(void* const* d_in, const int* in_sizes, int n_in,
                              void* d_out, int out_size, void* d_ws, size_t ws_size,
                              hipStream_t stream) {
  const float* x      = (const float*)d_in[0];
  const float* W_in   = (const float*)d_in[1];
  const float* b_in   = (const float*)d_in[2];
  const float* pos    = (const float*)d_in[3];
  const float* Wc     = (const float*)d_in[4];
  const float* Wk     = (const float*)d_in[5];
  const float* Wv     = (const float*)d_in[6];
  const float* Wq     = (const float*)d_in[7];
  const float* Wo     = (const float*)d_in[8];
  const float* g1     = (const float*)d_in[9];
  const float* be1    = (const float*)d_in[10];
  const float* Wr     = (const float*)d_in[11];
  const float* br     = (const float*)d_in[12];
  const float* W1     = (const float*)d_in[13];
  const float* bE1    = (const float*)d_in[14];
  const float* W2     = (const float*)d_in[15];
  const float* bE2    = (const float*)d_in[16];
  const float* g2     = (const float*)d_in[17];
  const float* be2    = (const float*)d_in[18];
  const float* g_out  = (const float*)d_in[19];
  const float* b_out  = (const float*)d_in[20];
  const float* W_head = (const float*)d_in[21];
  const float* b_head = (const float*)d_in[22];
  float* out = (float*)d_out;

  float* ws  = (float*)d_ws;
  float* h   = ws;                        // NT*128
  float* lat = h   + (size_t)NT * DDIM;   // NT*32
  float* Kb  = lat + (size_t)NT * 32;     // NT*128
  float* Vb  = Kb  + (size_t)NT * DDIM;   // NT*128
  float* Qb  = Vb  + (size_t)NT * DDIM;   // NT*128
  float* at  = Qb  + (size_t)NT * DDIM;   // NT*128
  float* tp  = at  + (size_t)NT * DDIM;   // NT*128
  float* h1  = tp  + (size_t)NT * DDIM;   // NT*256
  float* rw  = h1  + (size_t)NT * EDIM;   // NT*4

  const dim3 blk(128);
  const int rtiles = (NT / 16) / 4;   // 1024 blocks of 4 waves

  // embed: h = x @ W_in + b_in ; h += pos_emb
  gemm_wmma<4><<<dim3(rtiles, 2), blk, 0, stream>>>(x, W_in, b_in, nullptr, 0, h,
                                                    32, DDIM, 0, 0);
  pos_add<<<NT, blk, 0, stream>>>(h, pos);

  for (int l = 0; l < NLAY; ++l) {
    const float* Wc_l = Wc + (size_t)l * DDIM * 32;
    const float* Wk_l = Wk + (size_t)l * 32 * DDIM;
    const float* Wv_l = Wv + (size_t)l * 32 * DDIM;
    const float* Wq_l = Wq + (size_t)l * DDIM * DDIM;
    const float* Wo_l = Wo + (size_t)l * DDIM * DDIM;

    gemm_wmma<2><<<dim3(rtiles, 1), blk, 0, stream>>>(h,   Wc_l, nullptr, nullptr, 0, lat,
                                                      DDIM, 32, 0, 0);
    gemm_wmma<4><<<dim3(rtiles, 2), blk, 0, stream>>>(lat, Wk_l, nullptr, nullptr, 0, Kb,
                                                      32, DDIM, 0, 0);
    gemm_wmma<4><<<dim3(rtiles, 2), blk, 0, stream>>>(lat, Wv_l, nullptr, nullptr, 0, Vb,
                                                      32, DDIM, 0, 0);
    gemm_wmma<4><<<dim3(rtiles, 2), blk, 0, stream>>>(h,   Wq_l, nullptr, nullptr, 0, Qb,
                                                      DDIM, DDIM, 0, 0);
    mla_attn<<<dim3(4096), blk, 0, stream>>>(Qb, Kb, Vb, at);
    gemm_wmma<4><<<dim3(rtiles, 2), blk, 0, stream>>>(at, Wo_l, nullptr, nullptr, 0, tp,
                                                      DDIM, DDIM, 0, 0);
    add_ln<<<NT, blk, 0, stream>>>(h, tp, g1 + l * DDIM, be1 + l * DDIM, h);

    // MoE
    router_k<<<NT, blk, 0, stream>>>(h, Wr + (size_t)l * DDIM * NEXP, br + l * NEXP, rw);
    for (int e = 0; e < NEXP; ++e) {
      const float* W1_le  = W1  + (size_t)(l * NEXP + e) * DDIM * EDIM;
      const float* bE1_le = bE1 + (size_t)(l * NEXP + e) * EDIM;
      const float* W2_le  = W2  + (size_t)(l * NEXP + e) * EDIM * DDIM;
      const float* bE2_le = bE2 + (size_t)(l * NEXP + e) * DDIM;
      gemm_wmma<4><<<dim3(rtiles, 4), blk, 0, stream>>>(h, W1_le, bE1_le, nullptr, 0, h1,
                                                        DDIM, EDIM, 1, 0);
      gemm_wmma<4><<<dim3(rtiles, 2), blk, 0, stream>>>(h1, W2_le, bE2_le, rw + e, NEXP, tp,
                                                        EDIM, DDIM, 0, (e > 0) ? 1 : 0);
    }
    add_ln<<<NT, blk, 0, stream>>>(h, tp, g2 + l * DDIM, be2 + l * DDIM, h);
  }

  add_ln<<<NT, blk, 0, stream>>>(h, nullptr, g_out, b_out, tp);
  head_k<<<NT, blk, 0, stream>>>(tp, W_head, b_head, out);
}
